// ChunkedCausalAttention_StaticRoutedLoRAExpert_46737834115832
// MI455X (gfx1250) — compile-verified, matrix operand order checked by reading, not run on hardware
//
#include <hip/hip_runtime.h>
#include <hip/hip_bf16.h>

#define DIMSZ 768
#define NHEADS 12
#define HDIM 64
#define NCHUNK 4
#define CHUNK 1024
#define SEQLEN 4096
#define BATCH 2
#define NEXP 8
#define RLORA 64
#define LORA_SC 2.0f   /* 128/64 */

typedef __attribute__((ext_vector_type(16))) __bf16 v16bf;
typedef __attribute__((ext_vector_type(8)))  __bf16 v8bf;
typedef __attribute__((ext_vector_type(8)))  float  v8f;

__device__ __forceinline__ v8f wmma_bf16(v16bf a, v16bf b, v8f c) {
  return __builtin_amdgcn_wmma_f32_16x16x32_bf16(false, a, false, b, (short)0, c,
                                                 false, false);
}

// Load a 16x32 bf16 A/B fragment from a row-major bf16 tile (two b128 loads).
// prow = this lane's row pointer at k-offset 0 of the tile (16B-aligned).
// Layout (ISA 7.12.2): lanes 0-15 hold K 0-7 & 16-23, lanes 16-31 hold K 8-15 & 24-31.
__device__ __forceinline__ v16bf load_frag(const __bf16* __restrict__ prow, int lane) {
  const int kh = ((lane >> 4) & 1) * 8;
  v8bf lo = *(const v8bf*)(prow + kh);
  v8bf hi = *(const v8bf*)(prow + 16 + kh);
  v16bf f;
#pragma unroll
  for (int i = 0; i < 8; ++i) { f[i] = lo[i]; f[i + 8] = hi[i]; }
  return f;
}

// 32x64 output tile per wave: acc[0..3] = rows arow0[0:16], acc[4..7] = rows arow1.
// Every B fragment is reused by two WMMAs (8 WMMAs per 6 fragment loads).
__device__ __forceinline__ void gemm_acc2(v8f acc[8],
                                          const __bf16* __restrict__ arow0,
                                          const __bf16* __restrict__ arow1,
                                          const __bf16* __restrict__ W, int ldw,
                                          int wRow0, int K, int lane) {
  const int r = lane & 15;
  for (int k0 = 0; k0 < K; k0 += 32) {
    v16bf a0 = load_frag(arow0 + k0, lane);
    v16bf a1 = load_frag(arow1 + k0, lane);
#pragma unroll
    for (int n = 0; n < 4; ++n) {
      const __bf16* wr = W + (size_t)(wRow0 + 16 * n + r) * ldw + k0;
      if (k0 + 32 < K) __builtin_prefetch(wr + 32, 0, 0);  // global_prefetch_b8
      v16bf bf = load_frag(wr, lane);
      acc[n]     = wmma_bf16(a0, bf, acc[n]);
      acc[n + 4] = wmma_bf16(a1, bf, acc[n + 4]);
    }
  }
}

// -------- elementwise fp32 -> bf16 (8 elts/thread; all sizes are multiples of 8) ----
__global__ void cvt_bf16_kernel(const float* __restrict__ src,
                                __bf16* __restrict__ dst, long n) {
  const long i = ((long)blockIdx.x * blockDim.x + threadIdx.x) * 8;
  if (i + 8 <= n) {
#pragma unroll
    for (int j = 0; j < 8; ++j) dst[i + j] = (__bf16)src[i + j];
  }
}

// row m of the logical 2048-row activation block
__device__ __forceinline__ const __bf16* chunk_row(const __bf16* src, int m, int offs,
                                                   long batchStride) {
  return src + (size_t)(m >> 10) * batchStride + (size_t)(offs + (m & 1023)) * DIMSZ;
}

// -------- h = LORA_SC * (src_chunk @ Amat^T)   (M=2048, N=64, K=768), bf16 out ----
__global__ void lora_h_kernel(const __bf16* __restrict__ src,
                              const __bf16* __restrict__ Amat,
                              const long long* __restrict__ eidx, int chunk,
                              int offs, long batchStride,
                              __bf16* __restrict__ h) {
  const int e = (int)eidx[chunk];
  const __bf16* A = Amat + (size_t)e * RLORA * DIMSZ;
  const int wave = threadIdx.x >> 5, lane = threadIdx.x & 31;
  const int mBase = blockIdx.y * 128 + wave * 32;
  const int r = lane & 15;
  v8f acc[8] = {};
  gemm_acc2(acc, chunk_row(src, mBase + r, offs, batchStride),
            chunk_row(src, mBase + 16 + r, offs, batchStride), A, DIMSZ, 0, DIMSZ,
            lane);
  const int mhi = (lane >> 4) * 8;
#pragma unroll
  for (int g = 0; g < 2; ++g)
#pragma unroll
    for (int n = 0; n < 4; ++n) {
      const int col = 16 * n + (lane & 15);
#pragma unroll
      for (int j = 0; j < 8; ++j) {
        const int mm = mBase + 16 * g + mhi + j;
        h[(size_t)mm * RLORA + col] = (__bf16)(LORA_SC * acc[n + 4 * g][j]);
      }
    }
}

// -------- qkv = x@Wqkv^T + hs@Bflat^T ; scatter bf16 to Q(*scale) / Kcache / Vt ----
__global__ void qkv_kernel(const __bf16* __restrict__ x,
                           const __bf16* __restrict__ Wqkv,
                           const __bf16* __restrict__ Bq,
                           const __bf16* __restrict__ h,
                           const long long* __restrict__ eidx, int chunk, int offs,
                           __bf16* __restrict__ Qbuf, __bf16* __restrict__ Kc,
                           __bf16* __restrict__ Vt) {
  const int e = (int)eidx[chunk];
  const __bf16* Bfl = Bq + (size_t)e * (3 * DIMSZ) * RLORA;
  const int wave = threadIdx.x >> 5, lane = threadIdx.x & 31;
  const int mBase = blockIdx.y * 128 + wave * 32;
  const int colBase = blockIdx.x * 64;
  const int r = lane & 15;
  v8f acc[8] = {};
  gemm_acc2(acc, x + ((size_t)((mBase + r) >> 10) * SEQLEN + offs + ((mBase + r) & 1023)) * DIMSZ,
            x + ((size_t)((mBase + 16 + r) >> 10) * SEQLEN + offs + ((mBase + 16 + r) & 1023)) * DIMSZ,
            Wqkv, DIMSZ, colBase, DIMSZ, lane);
  gemm_acc2(acc, h + (size_t)(mBase + r) * RLORA, h + (size_t)(mBase + 16 + r) * RLORA,
            Bfl, RLORA, colBase, RLORA, lane);

  const int mhi = (lane >> 4) * 8;
#pragma unroll
  for (int n = 0; n < 4; ++n) {
    const int c = colBase + 16 * n + (lane & 15);
    const int t3 = c / DIMSZ;
    const int rem = c - t3 * DIMSZ;
    const int hh = rem >> 6, d = rem & 63;
#pragma unroll
    for (int g = 0; g < 2; ++g)
#pragma unroll
      for (int j = 0; j < 8; ++j) {
        const int mm = mBase + 16 * g + mhi + j;
        const int bb = mm >> 10, tt = mm & 1023;
        const int sp = offs + tt;
        const float val = acc[n + 4 * g][j];
        if (t3 == 0)        // softmax scale folded into stored Q
          Qbuf[(((size_t)bb * NHEADS + hh) * CHUNK + tt) * HDIM + d] =
              (__bf16)(val * 0.125f);
        else if (t3 == 1)
          Kc[(((size_t)bb * NHEADS + hh) * SEQLEN + sp) * HDIM + d] = (__bf16)val;
        else
          Vt[(((size_t)bb * NHEADS + hh) * HDIM + d) * SEQLEN + sp] = (__bf16)val;
      }
  }
}

// -------- flash attention over kvLen keys; O -> (B, CHUNK, DIMSZ) bf16 --------
__global__ void attn_kernel(const __bf16* __restrict__ Qbuf,
                            const __bf16* __restrict__ Kc,
                            const __bf16* __restrict__ Vt, int kvLen,
                            __bf16* __restrict__ O) {
  const int bh = blockIdx.x;               // b*NHEADS + h
  const int wave = threadIdx.x >> 5, lane = threadIdx.x & 31;
  const int qBase = blockIdx.y * 64 + wave * 16;
  const __bf16* Q = Qbuf + (size_t)bh * CHUNK * HDIM;
  const __bf16* K = Kc + (size_t)bh * SEQLEN * HDIM;
  const __bf16* V = Vt + (size_t)bh * HDIM * SEQLEN;
  const int r = lane & 15;
  v16bf qf0 = load_frag(Q + (size_t)(qBase + r) * HDIM + 0, lane);
  v16bf qf1 = load_frag(Q + (size_t)(qBase + r) * HDIM + 32, lane);

  v8f o[4] = {};
  float mrow[8], lrow[8];
#pragma unroll
  for (int j = 0; j < 8; ++j) { mrow[j] = -1e30f; lrow[j] = 0.0f; }

  __shared__ __align__(16) __bf16 Pb[4][16 * 32];
  __bf16* pt = Pb[wave];
  const int mhi = (lane >> 4) * 8;

  for (int s0 = 0; s0 < kvLen; s0 += 32) {
    // S = Q K^T for 32 keys -> two 16x16 C tiles
    v8f sacc[2] = {};
#pragma unroll
    for (int nt = 0; nt < 2; ++nt) {
      const __bf16* krow = K + (size_t)(s0 + 16 * nt + r) * HDIM;
      sacc[nt] = wmma_bf16(qf0, load_frag(krow + 0, lane), sacc[nt]);
      sacc[nt] = wmma_bf16(qf1, load_frag(krow + 32, lane), sacc[nt]);
    }
    // online softmax per query row (row lives across 16 lanes of a half-wave)
#pragma unroll
    for (int j = 0; j < 8; ++j) {
      float v0 = sacc[0][j], v1 = sacc[1][j];
      float tmax = fmaxf(v0, v1);
      for (int off = 1; off < 16; off <<= 1)
        tmax = fmaxf(tmax, __shfl_xor(tmax, off, 32));
      const float mnew = fmaxf(mrow[j], tmax);
      const float p0 = __expf(v0 - mnew), p1 = __expf(v1 - mnew);
      float psum = p0 + p1;
      for (int off = 1; off < 16; off <<= 1) psum += __shfl_xor(psum, off, 32);
      const float corr = __expf(mrow[j] - mnew);
      lrow[j] = lrow[j] * corr + psum;
      mrow[j] = mnew;
#pragma unroll
      for (int d = 0; d < 4; ++d) o[d][j] *= corr;
      sacc[0][j] = p0;
      sacc[1][j] = p1;
    }
    // C-layout -> A-fragment relayout through per-wave LDS (16x32 bf16)
#pragma unroll
    for (int nt = 0; nt < 2; ++nt)
#pragma unroll
      for (int j = 0; j < 8; ++j)
        pt[(mhi + j) * 32 + nt * 16 + (lane & 15)] = (__bf16)sacc[nt][j];
    __syncthreads();
    v16bf pf = load_frag(pt + r * 32, lane);
    // O += P * V  (V transposed cache: rows are head-dims, contiguous in keys)
#pragma unroll
    for (int d = 0; d < 4; ++d) {
      v16bf vf = load_frag(V + (size_t)(16 * d + r) * SEQLEN + s0, lane);
      o[d] = wmma_bf16(pf, vf, o[d]);
    }
    __syncthreads();
  }
  // normalize and write O as (B, CHUNK, DIMSZ) bf16
  const int bb = bh / NHEADS, hh = bh % NHEADS;
#pragma unroll
  for (int d = 0; d < 4; ++d) {
    const int dcol = 16 * d + (lane & 15);
#pragma unroll
    for (int j = 0; j < 8; ++j) {
      const int qq = qBase + mhi + j;
      O[((size_t)bb * CHUNK + qq) * DIMSZ + hh * HDIM + dcol] =
          (__bf16)(o[d][j] / lrow[j]);
    }
  }
}

// -------- out = O@Wproj^T + h2s@Bproj^T, written fp32 into d_out --------
__global__ void proj_kernel(const __bf16* __restrict__ O,
                            const __bf16* __restrict__ Wproj,
                            const __bf16* __restrict__ Bp,
                            const __bf16* __restrict__ h2,
                            const long long* __restrict__ eidx, int chunk, int offs,
                            float* __restrict__ out) {
  const int e = (int)eidx[chunk];
  const __bf16* Bfl = Bp + (size_t)e * DIMSZ * RLORA;
  const int wave = threadIdx.x >> 5, lane = threadIdx.x & 31;
  const int mBase = blockIdx.y * 128 + wave * 32;
  const int colBase = blockIdx.x * 64;
  const int r = lane & 15;
  v8f acc[8] = {};
  gemm_acc2(acc, O + (size_t)(mBase + r) * DIMSZ, O + (size_t)(mBase + 16 + r) * DIMSZ,
            Wproj, DIMSZ, colBase, DIMSZ, lane);
  gemm_acc2(acc, h2 + (size_t)(mBase + r) * RLORA,
            h2 + (size_t)(mBase + 16 + r) * RLORA, Bfl, RLORA, colBase, RLORA, lane);
  const int mhi = (lane >> 4) * 8;
#pragma unroll
  for (int n = 0; n < 4; ++n) {
    const int c = colBase + 16 * n + (lane & 15);
#pragma unroll
    for (int g = 0; g < 2; ++g)
#pragma unroll
      for (int j = 0; j < 8; ++j) {
        const int mm = mBase + 16 * g + mhi + j;
        const int bb = mm >> 10, tt = mm & 1023;
        out[((size_t)bb * SEQLEN + offs + tt) * DIMSZ + c] = acc[n + 4 * g][j];
      }
  }
}

static inline void cvt_launch(const float* s, __bf16* d, long n, hipStream_t st) {
  const long threads = n / 8;
  cvt_bf16_kernel<<<dim3((unsigned)((threads + 255) / 256)), dim3(256), 0, st>>>(s, d, n);
}

extern "C" void kernel_launch(void* const* d_in, const int* in_sizes, int n_in,
                              void* d_out, int out_size, void* d_ws, size_t ws_size,
                              hipStream_t stream) {
  (void)in_sizes; (void)n_in; (void)out_size; (void)ws_size;
  const float* x     = (const float*)d_in[0];
  const float* Wqkv  = (const float*)d_in[1];
  const float* Aqkv  = (const float*)d_in[2];
  const float* Bqkv  = (const float*)d_in[3];
  const float* Wproj = (const float*)d_in[4];
  const float* Aproj = (const float*)d_in[5];
  const float* Bproj = (const float*)d_in[6];
  const long long* eidx = (const long long*)d_in[7];
  float* out = (float*)d_out;

  const long nX  = (long)BATCH * SEQLEN * DIMSZ;
  const long nWq = (long)3 * DIMSZ * DIMSZ;
  const long nAq = (long)NEXP * RLORA * DIMSZ;
  const long nBq = (long)NEXP * 3 * DIMSZ * RLORA;
  const long nWp = (long)DIMSZ * DIMSZ;
  const long nAp = (long)NEXP * RLORA * DIMSZ;
  const long nBp = (long)NEXP * DIMSZ * RLORA;

  char* ws = (char*)d_ws;
  __bf16* xbf  = (__bf16*)ws; ws += nX * 2;
  __bf16* Wqb  = (__bf16*)ws; ws += nWq * 2;
  __bf16* Aqb  = (__bf16*)ws; ws += nAq * 2;
  __bf16* Bqb  = (__bf16*)ws; ws += nBq * 2;
  __bf16* Wpb  = (__bf16*)ws; ws += nWp * 2;
  __bf16* Apb  = (__bf16*)ws; ws += nAp * 2;
  __bf16* Bpb  = (__bf16*)ws; ws += nBp * 2;
  __bf16* Qbuf = (__bf16*)ws; ws += (size_t)BATCH * NHEADS * CHUNK * HDIM * 2;
  __bf16* Kc   = (__bf16*)ws; ws += (size_t)BATCH * NHEADS * SEQLEN * HDIM * 2;
  __bf16* Vt   = (__bf16*)ws; ws += (size_t)BATCH * NHEADS * SEQLEN * HDIM * 2;
  __bf16* hbuf = (__bf16*)ws; ws += (size_t)BATCH * CHUNK * RLORA * 2;
  __bf16* Obuf = (__bf16*)ws; ws += (size_t)BATCH * CHUNK * DIMSZ * 2;

  // One-time fp32 -> bf16 pre-conversion (weights + activations).
  cvt_launch(x, xbf, nX, stream);
  cvt_launch(Wqkv, Wqb, nWq, stream);
  cvt_launch(Aqkv, Aqb, nAq, stream);
  cvt_launch(Bqkv, Bqb, nBq, stream);
  cvt_launch(Wproj, Wpb, nWp, stream);
  cvt_launch(Aproj, Apb, nAp, stream);
  cvt_launch(Bproj, Bpb, nBp, stream);

  const dim3 blk(128);
  for (int c = 0; c < NCHUNK; ++c) {
    const int offs = c * CHUNK;
    const int kv = (c + 1) * CHUNK;
    lora_h_kernel<<<dim3(1, 16), blk, 0, stream>>>(xbf, Aqb, eidx, c, offs,
                                                   (long)SEQLEN * DIMSZ, hbuf);
    qkv_kernel<<<dim3(36, 16), blk, 0, stream>>>(xbf, Wqb, Bqb, hbuf, eidx, c, offs,
                                                 Qbuf, Kc, Vt);
    attn_kernel<<<dim3(BATCH * NHEADS, 16), blk, 0, stream>>>(Qbuf, Kc, Vt, kv, Obuf);
    lora_h_kernel<<<dim3(1, 16), blk, 0, stream>>>(Obuf, Apb, eidx, c, 0,
                                                   (long)CHUNK * DIMSZ, hbuf);
    proj_kernel<<<dim3(12, 16), blk, 0, stream>>>(Obuf, Wpb, Bpb, hbuf, eidx, c,
                                                  offs, out);
  }
}
